// AtomTransformer_47502338294325
// MI455X (gfx1250) — matrix operand order repaired and checked by reading, not compiled, not run on hardware
//
#include <hip/hip_runtime.h>
#include <hip/hip_bf16.h>

// ---------------------------------------------------------------------------
// AF3 atom-transformer for MI455X (gfx1250, wave32, WMMA).
// Block-sparse attention (n_queries=32, n_keys=128) + bf16 WMMA GEMMs.
// B matrices pre-swizzled into WMMA fragment order -> global_load_b128 direct.
// A slabs double-buffered through LDS with gfx1250 async global->LDS copies.
// ---------------------------------------------------------------------------

typedef __attribute__((ext_vector_type(16))) __bf16        v16bf;
typedef __attribute__((ext_vector_type(8)))  float         v8f;
typedef __attribute__((ext_vector_type(8)))  unsigned int  v8u;

#define NATOM 3072
#define CA    128
#define NBLK  96      // NATOM / 32

// gfx1250 async global->LDS path (probe-confirmed signature:
//   (int vector_size(16) AS(1)*, ... , imm, imm), completion via ASYNCcnt).
#if defined(__AMDGCN__) && \
    __has_builtin(__builtin_amdgcn_global_load_async_to_lds_b128) && \
    __has_builtin(__builtin_amdgcn_s_wait_asynccnt)
#define ASYNC_LDS 1
typedef int nvi4 __attribute__((vector_size(16)));          // native 4 x i32
typedef __attribute__((address_space(1))) nvi4 GV4;         // global
typedef __attribute__((address_space(3))) nvi4 LV4;         // LDS
#else
#define ASYNC_LDS 0
#endif

// ---------------- WMMA helpers (wave32, contiguous k order k = 16*half + s) ------------------

__device__ __forceinline__ v8f wmma_bf16(v16bf a, v16bf b, v8f c) {
  return __builtin_amdgcn_wmma_f32_16x16x32_bf16(
      false, a, false, b, (short)0, c, false, false);
}

__device__ __forceinline__ unsigned pack2bf(float a, float b) {
  unsigned short x = __builtin_bit_cast(unsigned short, (__bf16)a);
  unsigned short y = __builtin_bit_cast(unsigned short, (__bf16)b);
  return (unsigned)x | ((unsigned)y << 16);
}

// 8 packed bf16 (one int4) -> 8 floats.
__device__ __forceinline__ void unpack8bf(int4 v, float* f) {
  unsigned uu[4] = {(unsigned)v.x, (unsigned)v.y, (unsigned)v.z, (unsigned)v.w};
#pragma unroll
  for (int j = 0; j < 4; ++j) {
    f[2 * j]     = __builtin_bit_cast(float, uu[j] << 16);
    f[2 * j + 1] = __builtin_bit_cast(float, uu[j] & 0xffff0000u);
  }
}

// 16 contiguous bf16 -> fragment register (2 x 16B vector loads).
__device__ __forceinline__ v16bf load16(const __bf16* p) {
  int4 lo = *(const int4*)p;
  int4 hi = *(const int4*)(p + 8);
  v8u u;
  u[0] = lo.x; u[1] = lo.y; u[2] = lo.z; u[3] = lo.w;
  u[4] = hi.x; u[5] = hi.y; u[6] = hi.z; u[7] = hi.w;
  return __builtin_bit_cast(v16bf, u);
}

// A fragment 16x32: lane<16 -> M=lane, K 0..15; lane>=16 -> K 16..31
__device__ __forceinline__ v16bf load_a_frag(const __bf16* smem, int ld,
                                             int r0, int k0) {
  const int lane = threadIdx.x & 31;
  const int half = lane >> 4, row = lane & 15;
  const __bf16* pr = smem + (size_t)(r0 + row) * ld + k0 + half * 16;
  int4 lo = *(const int4*)pr;          // K = base+0..7   (pairs)
  int4 hi = *(const int4*)(pr + 8);    // K = base+8..15  (pairs)
  v8u u;
  u[0] = lo.x; u[1] = lo.y; u[2] = lo.z; u[3] = lo.w;
  u[4] = hi.x; u[5] = hi.y; u[6] = hi.z; u[7] = hi.w;
  return __builtin_bit_cast(v16bf, u);
}

// B fragment 32x16 from an n-major buffer: element (k,n) at smem[n*sN + kb + k],
// lane<16 -> K 0..15, lane>=16 -> K 16..31, N = lane&15 (+n0).
__device__ __forceinline__ v16bf load_b_nmajor(const __bf16* smem, int n0,
                                               int sN, int kb) {
  const int lane = threadIdx.x & 31;
  const int half = lane >> 4, l15 = lane & 15;
  return load16(smem + (size_t)(n0 + l15) * sN + kb + half * 16);
}

// ---------------- GEMM:  C(f32,MxN) = A(bf16,MxK) @ Bsw(bf16, swizzled) -----
// Bsw layout: ((nt*(K/32)+kc)*32 + lane)*16 + j ; lane = half*16 + (n&15),
// j = k&15, nt = n/16, kc = k/32, half = (k&31)>>4.
template <int K>
__global__ __launch_bounds__(256) void gemm_bf16(
    float* __restrict__ C, const __bf16* __restrict__ A,
    const __bf16* __restrict__ Bsw, int M, int N) {
  __shared__ __align__(32) __bf16 As[2][128 * 32];   // double-buffered A slabs
  constexpr int KC = K / 32;
  const int t = threadIdx.x, lane = t & 31, wid = t >> 5;
  const int bm0 = blockIdx.x * 128, bn0 = blockIdx.y * 64;
  const int wr0 = (wid & 3) * 32;
  const int nt0 = (bn0 >> 4) + (wid >> 2) * 2;
  const int arow = t >> 1, aoff = (t & 1) * 16;

  v8f acc00 = {}, acc01 = {}, acc10 = {}, acc11 = {};

#if ASYNC_LDS
  auto issue = [&](int kc) {
    const __bf16* ag = A + (size_t)(bm0 + arow) * K + kc * 32 + aoff;
    __bf16* sd = &As[kc & 1][arow * 32 + aoff];
    __builtin_amdgcn_global_load_async_to_lds_b128(
        (GV4*)(size_t)ag, (LV4*)(unsigned)(size_t)sd, 0, 0);
    __builtin_amdgcn_global_load_async_to_lds_b128(
        (GV4*)(size_t)(ag + 8), (LV4*)(unsigned)(size_t)(sd + 8), 0, 0);
  };
  issue(0);
#pragma unroll
  for (int kc = 0; kc < KC; ++kc) {
    if (kc + 1 < KC) {
      issue(kc + 1);                       // prefetch next slab (other buffer)
      __builtin_amdgcn_s_wait_asynccnt(2); // in-order: current slab landed
    } else {
      __builtin_amdgcn_s_wait_asynccnt(0);
    }
    __syncthreads();
    const __bf16* Ab = As[kc & 1];
    v16bf b0 = load16(Bsw + ((size_t)(nt0 * KC + kc) * 32 + lane) * 16);
    v16bf b1 = load16(Bsw + ((size_t)((nt0 + 1) * KC + kc) * 32 + lane) * 16);
    v16bf a0 = load_a_frag(Ab, 32, wr0, 0);
    v16bf a1 = load_a_frag(Ab, 32, wr0 + 16, 0);
    acc00 = wmma_bf16(a0, b0, acc00);
    acc01 = wmma_bf16(a0, b1, acc01);
    acc10 = wmma_bf16(a1, b0, acc10);
    acc11 = wmma_bf16(a1, b1, acc11);
    __syncthreads();
  }
#else
#pragma unroll
  for (int kc = 0; kc < KC; ++kc) {
    const __bf16* ag = A + (size_t)(bm0 + arow) * K + kc * 32 + aoff;
    __bf16* sd = &As[0][arow * 32 + aoff];
    int4 v0 = *(const int4*)ag;
    int4 v1 = *(const int4*)(ag + 8);
    *(int4*)sd = v0;
    *(int4*)(sd + 8) = v1;
    __syncthreads();
    v16bf b0 = load16(Bsw + ((size_t)(nt0 * KC + kc) * 32 + lane) * 16);
    v16bf b1 = load16(Bsw + ((size_t)((nt0 + 1) * KC + kc) * 32 + lane) * 16);
    v16bf a0 = load_a_frag(As[0], 32, wr0, 0);
    v16bf a1 = load_a_frag(As[0], 32, wr0 + 16, 0);
    acc00 = wmma_bf16(a0, b0, acc00);
    acc01 = wmma_bf16(a0, b1, acc01);
    acc10 = wmma_bf16(a1, b0, acc10);
    acc11 = wmma_bf16(a1, b1, acc11);
    __syncthreads();
  }
#endif

  const int half = lane >> 4, l15 = lane & 15;
  const int wc0 = (wid >> 2) * 32;
#pragma unroll
  for (int v = 0; v < 8; ++v) {
    int r0g = bm0 + wr0 + half * 8 + v;
    int c0g = bn0 + wc0 + l15;
    C[(size_t)r0g * N + c0g]             = acc00[v];
    C[(size_t)r0g * N + c0g + 16]        = acc01[v];
    C[(size_t)(r0g + 16) * N + c0g]      = acc10[v];
    C[(size_t)(r0g + 16) * N + c0g + 16] = acc11[v];
  }
}

// ---------------- Row LayerNorm over 128 channels ---------------------------
__global__ __launch_bounds__(256) void ln_rows(
    float* __restrict__ outf, __bf16* __restrict__ outb,
    const float* __restrict__ in, const float* __restrict__ w,
    const float* __restrict__ b, int M) {
  const int wid = threadIdx.x >> 5, lane = threadIdx.x & 31;
  const int row = blockIdx.x * 8 + wid;
  if (row >= M) return;
  const float4 x = *(const float4*)(in + (size_t)row * CA + lane * 4);
  float s = x.x + x.y + x.z + x.w;
  float q = x.x * x.x + x.y * x.y + x.z * x.z + x.w * x.w;
#pragma unroll
  for (int m = 1; m < 32; m <<= 1) {
    s += __shfl_xor(s, m, 32);
    q += __shfl_xor(q, m, 32);
  }
  const float mean = s * (1.f / CA);
  const float var  = q * (1.f / CA) - mean * mean;
  const float rstd = rsqrtf(var + 1e-5f);
  float y[4] = {(x.x - mean) * rstd, (x.y - mean) * rstd,
                (x.z - mean) * rstd, (x.w - mean) * rstd};
  if (w) {
    const float4 ww = *(const float4*)(w + lane * 4);
    const float4 bb = *(const float4*)(b + lane * 4);
    y[0] = y[0] * ww.x + bb.x; y[1] = y[1] * ww.y + bb.y;
    y[2] = y[2] * ww.z + bb.z; y[3] = y[3] * ww.w + bb.w;
  }
  if (outf)
    *(float4*)(outf + (size_t)row * CA + lane * 4) =
        make_float4(y[0], y[1], y[2], y[3]);
  if (outb) {
    int2 o = make_int2(pack2bf(y[0], y[1]), pack2bf(y[2], y[3]));
    *(int2*)(outb + (size_t)row * CA + lane * 4) = o;
  }
}

// ---------------- Pack kernels ----------------------------------------------
__global__ void pack_bf16(__bf16* __restrict__ dst, const float* __restrict__ src,
                          int total) {
  for (int i = blockIdx.x * 256 + threadIdx.x; i < total; i += gridDim.x * 256)
    dst[i] = (__bf16)src[i];
}

// f32 [K,Ns] slice -> WMMA-fragment-swizzled bf16 at logical column offset.
__global__ void pack_sw(__bf16* __restrict__ dst, const float* __restrict__ src,
                        int K, int Ns, int colOff, int total) {
  for (int i = blockIdx.x * 256 + threadIdx.x; i < total; i += gridDim.x * 256) {
    int k = i / Ns, ns = i - k * Ns;
    int n = colOff + ns;
    int nt = n >> 4, l15 = n & 15;
    int kc = k >> 5, kr = k & 31;
    int half = kr >> 4, j = kr & 15;
    int lane = half * 16 + l15;
    size_t idx = (((size_t)nt * (K >> 5) + kc) * 32 + lane) * 16 + j;
    dst[idx] = (__bf16)src[i];
  }
}

// ---------------- Elementwise fusions ---------------------------------------
__global__ void adaln_combine(__bf16* __restrict__ out, const float* __restrict__ tmp,
                              const float* __restrict__ gb,
                              const float* __restrict__ an, int total) {
  int i = blockIdx.x * 256 + threadIdx.x;
  if (i >= total) return;
  int n = i & 127, m = i >> 7;
  float g  = tmp[(size_t)m * 256 + n] + gb[n];
  float sh = tmp[(size_t)m * 256 + 128 + n];
  float sg = 1.f / (1.f + __expf(-g));
  out[i] = (__bf16)(sg * an[i] + sh);
}

__global__ void gate_mul_add(float* __restrict__ out, const float* __restrict__ G,
                             const float* __restrict__ gb,
                             const float* __restrict__ X,
                             const float* __restrict__ add, int total) {
  int i = blockIdx.x * 256 + threadIdx.x;
  if (i >= total) return;
  int n = i & 127;
  float sg = 1.f / (1.f + __expf(-(G[i] + gb[n])));
  out[i] = (add ? add[i] : 0.f) + sg * X[i];
}

__global__ void silu_mul(__bf16* __restrict__ out, const float* __restrict__ H,
                         int total) {
  int i = blockIdx.x * 256 + threadIdx.x;
  if (i >= total) return;
  int m = i >> 8, j = i & 255;
  float x = H[(size_t)m * 512 + j];
  float y = H[(size_t)m * 512 + 256 + j];
  out[i] = (__bf16)((x / (1.f + __expf(-x))) * y);
}

// ---------------- Fused block-sparse attention ------------------------------
// One workgroup per 32-query block; key window [32b-48, 32b+79].
// qkvg row layout: [q(128)|k(128)|v(128)|gate(128)], head-major 4x32.
__global__ __launch_bounds__(256) void attn_kernel(
    __bf16* __restrict__ O, const float* __restrict__ qkvg,
    const float* __restrict__ plm, const float* __restrict__ amask,
    const float* __restrict__ zln_w, const float* __restrict__ zln_b,
    const float* __restrict__ zb_w, const float* __restrict__ q_b) {
  __shared__ __align__(32) __bf16 biasT[4 * 128 * 32];  // 32KB [h][k][q] (q contiguous)
  __shared__ __align__(32) __bf16 KVs[128 * 32];        // 8KB: K[key][c] then Vt[c][key]
  __shared__ __align__(32) __bf16 QGs[32 * 32];         // 2KB: Q[row][c] then gate^T[c][row]
  __shared__ __align__(32) __bf16 Ps[32 * 128];         // 8KB probs
  __shared__ __align__(32) __bf16 Os[32 * 128];         // 8KB gated output tile
  __shared__ float red[32 * 4];

  const int t  = threadIdx.x;
  const int q0 = blockIdx.x * 32;
  const int w0 = q0 - 48;

  // ---- z_bias: LN over 16 pair channels + @zb_w, into biasT[h][k][q] ----
  // Thread -> (k, quad of q): vector stores along q.
#pragma unroll 1
  for (int g = 0; g < 4; ++g) {
    int gid = t + 256 * g;               // 0..1023
    int k  = gid >> 3;                   // 0..127
    int qb = (gid & 7) * 4;              // 0,4,...,28
    int kg = w0 + k;
    float bh[4][4];                      // [qq][h]
    if (kg < 0 || kg >= NATOM) {
#pragma unroll
      for (int qq = 0; qq < 4; ++qq)
#pragma unroll
        for (int h = 0; h < 4; ++h) bh[qq][h] = -1e9f;
    } else {
      float mterm = (amask[kg] - 1.f) * 1e9f;
#pragma unroll
      for (int qq = 0; qq < 4; ++qq) {
        const float4* zp = (const float4*)(
            plm + ((size_t)(q0 + qb + qq) * NATOM + (size_t)kg) * 16);
        float4 z0 = zp[0], z1 = zp[1], z2 = zp[2], z3 = zp[3];
        float x[16] = {z0.x, z0.y, z0.z, z0.w, z1.x, z1.y, z1.z, z1.w,
                       z2.x, z2.y, z2.z, z2.w, z3.x, z3.y, z3.z, z3.w};
        float mean = 0.f;
#pragma unroll
        for (int c = 0; c < 16; ++c) mean += x[c];
        mean *= (1.f / 16.f);
        float var = 0.f;
#pragma unroll
        for (int c = 0; c < 16; ++c) { float d = x[c] - mean; var += d * d; }
        var *= (1.f / 16.f);
        float rstd = rsqrtf(var + 1e-5f);
        bh[qq][0] = bh[qq][1] = bh[qq][2] = bh[qq][3] = mterm;
#pragma unroll
        for (int c = 0; c < 16; ++c) {
          float y = (x[c] - mean) * rstd * zln_w[c] + zln_b[c];
#pragma unroll
          for (int h = 0; h < 4; ++h) bh[qq][h] += y * zb_w[c * 4 + h];
        }
      }
    }
#pragma unroll
    for (int h = 0; h < 4; ++h) {
      int2 o = make_int2(pack2bf(bh[0][h], bh[1][h]),
                         pack2bf(bh[2][h], bh[3][h]));
      *(int2*)&biasT[h * 4096 + k * 32 + qb] = o;
    }
  }
  __syncthreads();

  const int lane = t & 31, wid = t >> 5;
  const int half = lane >> 4, l15 = lane & 15;
  const float scale = 0.17677669529663687f;  // 1/sqrt(32)

  for (int h = 0; h < 4; ++h) {
    // ---- stage Q (+bias): thread -> row t>>3, 4 channels ----
    {
      int row = t >> 3, c0 = (t & 7) * 4;
      float4 x = *(const float4*)(qkvg + (size_t)(q0 + row) * 512 + h * 32 + c0);
      float4 b = *(const float4*)(q_b + h * 32 + c0);
      int2 o = make_int2(pack2bf(x.x + b.x, x.y + b.y),
                         pack2bf(x.z + b.z, x.w + b.w));
      *(int2*)&QGs[row * 32 + c0] = o;
    }
    // ---- stage K row-major [key][c]: thread -> key t>>1, 16 channels ----
    {
      int key = t >> 1, c0 = (t & 1) * 16;
      int kg = w0 + key;
      int4 o0 = make_int4(0, 0, 0, 0), o1 = make_int4(0, 0, 0, 0);
      if (kg >= 0 && kg < NATOM) {
        const float4* p = (const float4*)(qkvg + (size_t)kg * 512 + 128 + h * 32 + c0);
        float4 f0 = p[0], f1 = p[1], f2 = p[2], f3 = p[3];
        o0 = make_int4(pack2bf(f0.x, f0.y), pack2bf(f0.z, f0.w),
                       pack2bf(f1.x, f1.y), pack2bf(f1.z, f1.w));
        o1 = make_int4(pack2bf(f2.x, f2.y), pack2bf(f2.z, f2.w),
                       pack2bf(f3.x, f3.y), pack2bf(f3.z, f3.w));
      }
      *(int4*)&KVs[key * 32 + c0]     = o0;
      *(int4*)&KVs[key * 32 + c0 + 8] = o1;
    }
    __syncthreads();

    // ---- logits S[32,128] = Q @ K^T (K-frag is contiguous per lane) ----
    const int rt = wid >> 2, cp = wid & 3;
    v16bf af  = load_a_frag(QGs, 32, rt * 16, 0);
    v16bf bk0 = load_b_nmajor(KVs, cp * 32, 32, 0);       // n=key, k=channel
    v16bf bk1 = load_b_nmajor(KVs, cp * 32 + 16, 32, 0);
    v8f z0 = {}, z1 = {};
    v8f c0 = wmma_bf16(af, bk0, z0);
    v8f c1 = wmma_bf16(af, bk1, z1);

    const int rowBase = rt * 16 + half * 8;
    const int col0 = cp * 32 + l15;
    float bf0[8], bf1[8];
    unpack8bf(*(const int4*)&biasT[h * 4096 + col0 * 32 + rowBase], bf0);
    unpack8bf(*(const int4*)&biasT[h * 4096 + (col0 + 16) * 32 + rowBase], bf1);
    float l0[8], l1[8];
#pragma unroll
    for (int v = 0; v < 8; ++v) {
      l0[v] = c0[v] * scale + bf0[v];
      l1[v] = c1[v] * scale + bf1[v];
    }
    // ---- softmax: width-16 shfl + cross-wave via tiny LDS ----
    float pm[8];
#pragma unroll
    for (int v = 0; v < 8; ++v) {
      pm[v] = fmaxf(l0[v], l1[v]);
#pragma unroll
      for (int m = 1; m < 16; m <<= 1) pm[v] = fmaxf(pm[v], __shfl_xor(pm[v], m, 16));
    }
    if (l15 == 0) {
#pragma unroll
      for (int v = 0; v < 8; ++v) red[(rowBase + v) * 4 + cp] = pm[v];
    }
    __syncthreads();
    float rmax[8];
#pragma unroll
    for (int v = 0; v < 8; ++v) {
      const float* rr = &red[(rowBase + v) * 4];
      rmax[v] = fmaxf(fmaxf(rr[0], rr[1]), fmaxf(rr[2], rr[3]));
    }
    __syncthreads();
    float ps[8];
#pragma unroll
    for (int v = 0; v < 8; ++v) {
      l0[v] = __expf(l0[v] - rmax[v]);
      l1[v] = __expf(l1[v] - rmax[v]);
      ps[v] = l0[v] + l1[v];
#pragma unroll
      for (int m = 1; m < 16; m <<= 1) ps[v] += __shfl_xor(ps[v], m, 16);
    }
    if (l15 == 0) {
#pragma unroll
      for (int v = 0; v < 8; ++v) red[(rowBase + v) * 4 + cp] = ps[v];
    }
    __syncthreads();
#pragma unroll
    for (int v = 0; v < 8; ++v) {
      const float* rr = &red[(rowBase + v) * 4];
      float inv = 1.f / (rr[0] + rr[1] + rr[2] + rr[3]);
      int row = rowBase + v;
      Ps[row * 128 + col0]      = (__bf16)(l0[v] * inv);
      Ps[row * 128 + col0 + 16] = (__bf16)(l1[v] * inv);
    }
    __syncthreads();

    // ---- stage V transposed Vt[c][key] + gate^T into QGs (Q consumed) ----
    {
      int c = t & 31, kb = (t >> 5) * 16;
      float vv[16];
#pragma unroll
      for (int kk = 0; kk < 16; ++kk) {
        int kg = w0 + kb + kk;
        vv[kk] = (kg >= 0 && kg < NATOM)
                     ? qkvg[(size_t)kg * 512 + 256 + h * 32 + c] : 0.f;
      }
      int4 o0 = make_int4(pack2bf(vv[0], vv[1]), pack2bf(vv[2], vv[3]),
                          pack2bf(vv[4], vv[5]), pack2bf(vv[6], vv[7]));
      int4 o1 = make_int4(pack2bf(vv[8], vv[9]), pack2bf(vv[10], vv[11]),
                          pack2bf(vv[12], vv[13]), pack2bf(vv[14], vv[15]));
      *(int4*)&KVs[c * 128 + kb]     = o0;
      *(int4*)&KVs[c * 128 + kb + 8] = o1;

      int r4 = (t >> 5) * 4;
      float g0 = qkvg[(size_t)(q0 + r4 + 0) * 512 + 384 + h * 32 + c];
      float g1 = qkvg[(size_t)(q0 + r4 + 1) * 512 + 384 + h * 32 + c];
      float g2 = qkvg[(size_t)(q0 + r4 + 2) * 512 + 384 + h * 32 + c];
      float g3 = qkvg[(size_t)(q0 + r4 + 3) * 512 + 384 + h * 32 + c];
      int2 og = make_int2(pack2bf(g0, g1), pack2bf(g2, g3));
      *(int2*)&QGs[c * 32 + r4] = og;    // gate^T[c][row]
    }
    __syncthreads();

    // ---- O[32,32] = P[32,128] @ V[128,32]; waves 0..3, gated epilogue ----
    if (wid < 4) {
      const int rt2 = wid >> 1, ct2 = wid & 1;
      v8f acc = {};
#pragma unroll
      for (int kc = 0; kc < 4; ++kc) {
        v16bf ap = load_a_frag(Ps, 128, rt2 * 16, kc * 32);
        v16bf bv = load_b_nmajor(KVs, ct2 * 16, 128, kc * 32);  // n=c, k=key
        acc = wmma_bf16(ap, bv, acc);
      }
      const int rb = rt2 * 16 + half * 8;
      const int c = ct2 * 16 + l15;
      float gv[8];
      unpack8bf(*(const int4*)&QGs[c * 32 + rb], gv);
#pragma unroll
      for (int v = 0; v < 8; ++v) {
        float sg = 1.f / (1.f + __expf(-gv[v]));
        Os[(rb + v) * 128 + h * 32 + c] = (__bf16)(sg * acc[v]);
      }
    }
    __syncthreads();
  }

  // ---- coalesced write of the 32x128 gated output tile ----
  {
    int row = t >> 3, c0 = (t & 7) * 16;
    int4 a = *(const int4*)&Os[row * 128 + c0];
    int4 b = *(const int4*)&Os[row * 128 + c0 + 8];
    *(int4*)(O + (size_t)(q0 + row) * CA + c0)     = a;
    *(int4*)(O + (size_t)(q0 + row) * CA + c0 + 8) = b;
  }
}

// ---------------------------------------------------------------------------
extern "C" void kernel_launch(void* const* d_in, const int* in_sizes, int n_in,
                              void* d_out, int out_size, void* d_ws, size_t ws_size,
                              hipStream_t stream) {
  (void)in_sizes; (void)n_in; (void)out_size; (void)ws_size;
  const int M = NATOM;

  const float* ql        = (const float*)d_in[0];
  const float* cl        = (const float*)d_in[1];
  const float* plm       = (const float*)d_in[2];
  const float* amask     = (const float*)d_in[3];
  const float* att_sln_w = (const float*)d_in[4];
  const float* att_sln_b = (const float*)d_in[5];
  const float* att_g_w   = (const float*)d_in[6];
  const float* att_g_b   = (const float*)d_in[7];
  const float* att_sh_w  = (const float*)d_in[8];
  const float* att_sg_w  = (const float*)d_in[9];
  const float* att_sg_b  = (const float*)d_in[10];
  const float* tr_sln_w  = (const float*)d_in[11];
  const float* tr_sln_b  = (const float*)d_in[12];
  const float* tr_g_w    = (const float*)d_in[13];
  const float* tr_g_b    = (const float*)d_in[14];
  const float* tr_sh_w   = (const float*)d_in[15];
  const float* tr_sg_w   = (const float*)d_in[16];
  const float* tr_sg_b   = (const float*)d_in[17];
  const float* att_q_w   = (const float*)d_in[18];
  const float* att_q_b   = (const float*)d_in[19];
  const float* att_k_w   = (const float*)d_in[20];
  const float* att_v_w   = (const float*)d_in[21];
  const float* att_zln_w = (const float*)d_in[22];
  const float* att_zln_b = (const float*)d_in[23];
  const float* att_zb_w  = (const float*)d_in[24];
  const float* att_gate_w= (const float*)d_in[25];
  const float* att_o_w   = (const float*)d_in[26];
  const float* tr_w1     = (const float*)d_in[27];
  const float* tr_w2     = (const float*)d_in[28];
  const float* tr_wo     = (const float*)d_in[29];
  float* out = (float*)d_out;

  char* wp = (char*)d_ws;
  auto alloc = [&](size_t bytes) -> void* {
    void* r = wp;
    wp += (bytes + 255) & ~(size_t)255;
    return r;
  };
  __bf16* s_bf    = (__bf16*)alloc((size_t)M * 128 * 2);
  __bf16* Wgsh_a  = (__bf16*)alloc((size_t)3 * 128 * 256 * 2);
  __bf16* Wgsh_t  = (__bf16*)alloc((size_t)3 * 128 * 256 * 2);
  __bf16* Wqkvg   = (__bf16*)alloc((size_t)3 * 128 * 512 * 2);
  __bf16* W12     = (__bf16*)alloc((size_t)3 * 128 * 512 * 2);
  __bf16* Wo      = (__bf16*)alloc((size_t)3 * 128 * 128 * 2);
  __bf16* Wsga    = (__bf16*)alloc((size_t)3 * 128 * 128 * 2);
  __bf16* Wsgt    = (__bf16*)alloc((size_t)3 * 128 * 128 * 2);
  __bf16* Wwo     = (__bf16*)alloc((size_t)3 * 256 * 128 * 2);
  __bf16* sn_bf   = (__bf16*)alloc((size_t)M * 128 * 2);
  __bf16* aln_bf  = (__bf16*)alloc((size_t)M * 128 * 2);
  __bf16* O_bf    = (__bf16*)alloc((size_t)M * 128 * 2);
  __bf16* hid_bf  = (__bf16*)alloc((size_t)M * 256 * 2);
  float*  a_n     = (float*)alloc((size_t)M * 128 * 4);
  float*  aCur    = (float*)alloc((size_t)M * 128 * 4);
  float*  tmpA    = (float*)alloc((size_t)M * 256 * 4);
  float*  qkvgF   = (float*)alloc((size_t)M * 512 * 4);   // reused as h12
  float*  OpF     = (float*)alloc((size_t)M * 128 * 4);   // reused as T
  float*  sgF     = (float*)alloc((size_t)M * 128 * 4);
  float*  battF   = (float*)alloc((size_t)M * 128 * 4);

  auto packw = [&](__bf16* dst, const float* src, int K, int Ns, int colOff) {
    int total = K * Ns;
    pack_sw<<<dim3((total + 255) / 256), dim3(256), 0, stream>>>(
        dst, src, K, Ns, colOff, total);
  };

  // --- prep: fragment-swizzled bf16 weights (fused concats) + bf16 s ---
  for (int i = 0; i < 3; ++i) {
    packw(Wgsh_a + (size_t)i * 128 * 256, att_g_w  + (size_t)i * 128 * 128, 128, 128, 0);
    packw(Wgsh_a + (size_t)i * 128 * 256, att_sh_w + (size_t)i * 128 * 128, 128, 128, 128);
    packw(Wgsh_t + (size_t)i * 128 * 256, tr_g_w   + (size_t)i * 128 * 128, 128, 128, 0);
    packw(Wgsh_t + (size_t)i * 128 * 256, tr_sh_w  + (size_t)i * 128 * 128, 128, 128, 128);
    packw(Wqkvg + (size_t)i * 128 * 512, att_q_w    + (size_t)i * 128 * 128, 128, 128, 0);
    packw(Wqkvg + (size_t)i * 128 * 512, att_k_w    + (size_t)i * 128 * 128, 128, 128, 128);
    packw(Wqkvg + (size_t)i * 128 * 512, att_v_w    + (size_t)i * 128 * 128, 128, 128, 256);
    packw(Wqkvg + (size_t)i * 128 * 512, att_gate_w + (size_t)i * 128 * 128, 128, 128, 384);
    packw(W12 + (size_t)i * 128 * 512, tr_w1 + (size_t)i * 128 * 256, 128, 256, 0);
    packw(W12 + (size_t)i * 128 * 512, tr_w2 + (size_t)i * 128 * 256, 128, 256, 256);
    packw(Wo   + (size_t)i * 128 * 128, att_o_w  + (size_t)i * 128 * 128, 128, 128, 0);
    packw(Wsga + (size_t)i * 128 * 128, att_sg_w + (size_t)i * 128 * 128, 128, 128, 0);
    packw(Wsgt + (size_t)i * 128 * 128, tr_sg_w  + (size_t)i * 128 * 128, 128, 128, 0);
    packw(Wwo  + (size_t)i * 256 * 128, tr_wo    + (size_t)i * 256 * 128, 256, 128, 0);
  }
  pack_bf16<<<dim3((M * 128 + 255) / 256), dim3(256), 0, stream>>>(s_bf, cl, M * 128);

  const int EB = (M * 128 + 255) / 256;
  const float* aSrc = ql;

  for (int i = 0; i < 3; ++i) {
    const size_t o128 = (size_t)i * 128, o16 = (size_t)i * 16, o64 = (size_t)i * 64;
    // a_n = LN(a), fp32
    ln_rows<<<dim3(M / 8), dim3(256), 0, stream>>>(a_n, (__bf16*)nullptr, aSrc,
                                                   nullptr, nullptr, M);
    // s_n (att)
    ln_rows<<<dim3(M / 8), dim3(256), 0, stream>>>(nullptr, sn_bf, cl,
                                                   att_sln_w + o128, att_sln_b + o128, M);
    gemm_bf16<128><<<dim3(M / 128, 4), dim3(256), 0, stream>>>(
        tmpA, sn_bf, Wgsh_a + (size_t)i * 128 * 256, M, 256);
    adaln_combine<<<dim3(EB), dim3(256), 0, stream>>>(aln_bf, tmpA, att_g_b + o128,
                                                      a_n, M * 128);
    gemm_bf16<128><<<dim3(M / 128, 8), dim3(256), 0, stream>>>(
        qkvgF, aln_bf, Wqkvg + (size_t)i * 128 * 512, M, 512);
    attn_kernel<<<dim3(NBLK), dim3(256), 0, stream>>>(
        O_bf, qkvgF, plm, amask, att_zln_w + o16, att_zln_b + o16,
        att_zb_w + o64, att_q_b + o128);
    gemm_bf16<128><<<dim3(M / 128, 2), dim3(256), 0, stream>>>(
        OpF, O_bf, Wo + (size_t)i * 128 * 128, M, 128);
    gemm_bf16<128><<<dim3(M / 128, 2), dim3(256), 0, stream>>>(
        sgF, s_bf, Wsga + (size_t)i * 128 * 128, M, 128);
    gate_mul_add<<<dim3(EB), dim3(256), 0, stream>>>(battF, sgF, att_sg_b + o128,
                                                     OpF, nullptr, M * 128);
    // transition
    ln_rows<<<dim3(M / 8), dim3(256), 0, stream>>>(nullptr, sn_bf, cl,
                                                   tr_sln_w + o128, tr_sln_b + o128, M);
    gemm_bf16<128><<<dim3(M / 128, 4), dim3(256), 0, stream>>>(
        tmpA, sn_bf, Wgsh_t + (size_t)i * 128 * 256, M, 256);
    adaln_combine<<<dim3(EB), dim3(256), 0, stream>>>(aln_bf, tmpA, tr_g_b + o128,
                                                      a_n, M * 128);
    gemm_bf16<128><<<dim3(M / 128, 8), dim3(256), 0, stream>>>(
        qkvgF, aln_bf, W12 + (size_t)i * 128 * 512, M, 512);
    silu_mul<<<dim3((M * 256 + 255) / 256), dim3(256), 0, stream>>>(hid_bf, qkvgF,
                                                                    M * 256);
    gemm_bf16<256><<<dim3(M / 128, 2), dim3(256), 0, stream>>>(
        OpF, hid_bf, Wwo + (size_t)i * 256 * 128, M, 128);
    gemm_bf16<128><<<dim3(M / 128, 2), dim3(256), 0, stream>>>(
        sgF, s_bf, Wsgt + (size_t)i * 128 * 128, M, 128);
    float* dst = (i == 2) ? out : aCur;
    gate_mul_add<<<dim3(EB), dim3(256), 0, stream>>>(dst, sgF, tr_sg_b + o128,
                                                     OpF, battF, M * 128);
    aSrc = aCur;
  }
}
